// MultiHeadAttention_21595095564389
// MI455X (gfx1250) — matrix operand order repaired and checked by reading, not compiled, not run on hardware
//
#include <hip/hip_runtime.h>
#include <hip/hip_bf16.h>

typedef __attribute__((ext_vector_type(16))) _Float16 v16h;
typedef __attribute__((ext_vector_type(8)))  float    v8f;

union HFragU { uint4 u4[2]; v16h h; };

// A-fragment (16x32 f16, MxK): lane L holds row M = L%16.
// Elements 0..7  = K kb..kb+7, elements 8..15 = K kb+16..kb+23,
// where kb = (L/16)*8 relative to the 32-wide K slice.  (ISA 7.12.2)
__device__ inline v16h load_a_frag(const _Float16* __restrict__ base, int ld, int row, int kb) {
    HFragU u;
    u.u4[0] = *reinterpret_cast<const uint4*>(base + (size_t)row * ld + kb);
    u.u4[1] = *reinterpret_cast<const uint4*>(base + (size_t)row * ld + kb + 16);
    return u.h;
}

// B-fragment (32x16 f16, KxN): lane L holds col N = L%16.
// Elements 0..7 = K kb..kb+7, elements 8..15 = K kb+16..kb+23, kb = (L/16)*8. (ISA 7.12.2)
__device__ inline v16h load_b_frag(const _Float16* __restrict__ base, int ld, int nrow, int kb) {
    HFragU u;
    const uint4* p = reinterpret_cast<const uint4*>(base + (size_t)nrow * ld + kb);
    u.u4[0] = p[0];
    u.u4[1] = p[2];
    return u.h;
}

// ---------------------------------------------------------------------------
// Kernel 1: x (B,C,H,W) -> x_seq (B,seq,d) + 2D positional encoding.
// fp32 copy kept for residual, f16 copy for WMMA GEMMs.
// ---------------------------------------------------------------------------
__global__ __launch_bounds__(256) void prep_x_kernel(const float* __restrict__ x,
                                                     float* __restrict__ Xf,
                                                     _Float16* __restrict__ Xh) {
    int idx = blockIdx.x * 256 + threadIdx.x;       // 8*1024*512 = 4194304 exact
    int c = idx & 511;
    int s = (idx >> 9) & 1023;
    int b = idx >> 19;
    float xv = x[((size_t)(b * 512 + c)) * 1024 + s];
    int i = c >> 2, q = c & 3;
    // div[i] = exp(2i * -ln(10000)/512)
    float freq = __expf(-(float)(2 * i) * (9.210340371976184f / 512.0f));
    float pos = (q < 2) ? (float)(s >> 5) : (float)(s & 31);   // pos_h : pos_w
    float ang = pos * freq;
    float pe = (q & 1) ? __cosf(ang) : __sinf(ang);
    float y = xv + pe;
    Xf[idx] = y;
    Xh[idx] = (_Float16)y;
}

// ---------------------------------------------------------------------------
// Kernel 2: convert the four 512x512 weight matrices to f16.
// ---------------------------------------------------------------------------
__global__ __launch_bounds__(256) void conv_w_kernel(const float* __restrict__ a, const float* __restrict__ b,
                                                     const float* __restrict__ c, const float* __restrict__ d,
                                                     _Float16* __restrict__ ah, _Float16* __restrict__ bh,
                                                     _Float16* __restrict__ ch, _Float16* __restrict__ dh) {
    int i = blockIdx.x * 256 + threadIdx.x;         // 512*512 = 262144 exact
    ah[i] = (_Float16)a[i];
    bh[i] = (_Float16)b[i];
    ch[i] = (_Float16)c[i];
    dh[i] = (_Float16)d[i];
}

// ---------------------------------------------------------------------------
// Kernel 3: NT GEMM  C[m,n] = sum_k A[m,k] * B[n,k]   (A: MxK, B: NxK, f16)
// Block = 8 waves, wave tile 32x64, block tile 128x128, all v_wmma_f32_16x16x32_f16.
// ---------------------------------------------------------------------------
template <bool F16OUT>
__global__ __launch_bounds__(256) void gemm_nt_kernel(const _Float16* __restrict__ A,
                                                      const _Float16* __restrict__ B,
                                                      void* __restrict__ Cout,
                                                      int N, int K) {
    const int lane = threadIdx.x & 31;
    const int wid  = threadIdx.x >> 5;
    const int lm = lane & 15, lh = lane >> 4;
    const int rowBase = blockIdx.x * 128 + (wid & 3) * 32;
    const int colBase = blockIdx.y * 128 + (wid >> 2) * 64;

    v8f zero = {};
    v8f c[2][4];
#pragma unroll
    for (int i = 0; i < 2; i++)
#pragma unroll
        for (int j = 0; j < 4; j++) c[i][j] = zero;

    for (int k0 = 0; k0 < K; k0 += 32) {
        v16h a[2], bf[4];
#pragma unroll
        for (int i = 0; i < 2; i++)
            a[i] = load_a_frag(A, K, rowBase + i * 16 + lm, k0 + lh * 8);
#pragma unroll
        for (int j = 0; j < 4; j++)
            bf[j] = load_b_frag(B, K, colBase + j * 16 + lm, k0 + lh * 8);
#pragma unroll
        for (int i = 0; i < 2; i++)
#pragma unroll
            for (int j = 0; j < 4; j++)
                c[i][j] = __builtin_amdgcn_wmma_f32_16x16x32_f16(
                    false, a[i], false, bf[j], (short)0, c[i][j], false, false);
    }

#pragma unroll
    for (int i = 0; i < 2; i++)
#pragma unroll
        for (int j = 0; j < 4; j++)
#pragma unroll
            for (int v = 0; v < 8; v++) {
                int r  = rowBase + i * 16 + v + lh * 8;
                int cc = colBase + j * 16 + lm;
                if (F16OUT)
                    ((_Float16*)Cout)[(size_t)r * N + cc] = (_Float16)c[i][j][v];
                else
                    ((float*)Cout)[(size_t)r * N + cc] = c[i][j][v];
            }
}

// ---------------------------------------------------------------------------
// Kernel 4: V (B,seq,d) -> Vt (B,h,dk,seq) so P@V B-frags read contiguous keys.
// ---------------------------------------------------------------------------
__global__ __launch_bounds__(256) void transpose_v_kernel(const _Float16* __restrict__ V,
                                                          _Float16* __restrict__ Vt) {
    int idx = blockIdx.x * 256 + threadIdx.x;       // 8192*512 = 4194304 exact
    int s  = idx & 1023;
    int d  = (idx >> 10) & 63;
    int bh = idx >> 16;
    int b = bh >> 3, h = bh & 7;
    Vt[idx] = V[((size_t)(b * 1024 + s)) * 512 + h * 64 + d];
}

// ---------------------------------------------------------------------------
// Kernel 5: flash attention.  Grid (seq/128, B*nh), block 8 waves.
// Wave owns 16 query rows; loops over 16 chunks of 64 keys.
// ---------------------------------------------------------------------------
__global__ __launch_bounds__(256) void attn_kernel(const _Float16* __restrict__ Q,
                                                   const _Float16* __restrict__ Kmat,
                                                   const _Float16* __restrict__ Vt,
                                                   _Float16* __restrict__ Out) {
    __shared__ _Float16 plds[8 * 16 * 64];          // 16 KB, per-wave P staging
    const int lane = threadIdx.x & 31;
    const int wid  = threadIdx.x >> 5;
    const int lm = lane & 15, lh = lane >> 4;
    const int bh = blockIdx.y;
    const int b = bh >> 3, h = bh & 7;
    const int qbase = blockIdx.x * 128 + wid * 16;

    const _Float16* Qp = Q    + (size_t)b * 1024 * 512 + h * 64;
    const _Float16* Kp = Kmat + (size_t)b * 1024 * 512 + h * 64;
    const _Float16* Vp = Vt   + (size_t)bh * 64 * 1024;
    _Float16* lw = plds + wid * 1024;

    v16h aq[2];
#pragma unroll
    for (int i = 0; i < 2; i++)
        aq[i] = load_a_frag(Qp, 512, qbase + lm, i * 32 + lh * 8);

    v8f zero = {};
    v8f o[4];
#pragma unroll
    for (int j = 0; j < 4; j++) o[j] = zero;
    float m8[8], l8[8];
#pragma unroll
    for (int v = 0; v < 8; v++) { m8[v] = -1e30f; l8[v] = 0.0f; }

    for (int kc = 0; kc < 16; kc++) {
        // S = Q @ K^T for this 16x64 tile
        v8f s[4];
#pragma unroll
        for (int n = 0; n < 4; n++) s[n] = zero;
#pragma unroll
        for (int ks = 0; ks < 2; ks++)
#pragma unroll
            for (int n = 0; n < 4; n++) {
                v16h bk = load_b_frag(Kp, 512, kc * 64 + n * 16 + lm, ks * 32 + lh * 8);
                s[n] = __builtin_amdgcn_wmma_f32_16x16x32_f16(
                    false, aq[ks], false, bk, (short)0, s[n], false, false);
            }

        // online softmax (rows live across 16 lanes of the same half)
        float p[4][8];
#pragma unroll
        for (int v = 0; v < 8; v++) {
            float r = -1e30f;
#pragma unroll
            for (int n = 0; n < 4; n++) {
                float sv = s[n][v] * 0.125f;        // 1/sqrt(dk)
                p[n][v] = sv;
                r = fmaxf(r, sv);
            }
#pragma unroll
            for (int msk = 1; msk < 16; msk <<= 1) r = fmaxf(r, __shfl_xor(r, msk, 32));
            float mn = fmaxf(m8[v], r);
            float al = __expf(m8[v] - mn);
            m8[v] = mn;
            float ls = 0.0f;
#pragma unroll
            for (int n = 0; n < 4; n++) {
                float e = __expf(p[n][v] - mn);
                p[n][v] = e;
                ls += e;
            }
#pragma unroll
            for (int msk = 1; msk < 16; msk <<= 1) ls += __shfl_xor(ls, msk, 32);
            l8[v] = l8[v] * al + ls;
#pragma unroll
            for (int j = 0; j < 4; j++) o[j][v] *= al;
        }

        // stage P through LDS to re-swizzle C-frag layout -> A-frag layout
#pragma unroll
        for (int n = 0; n < 4; n++)
#pragma unroll
            for (int v = 0; v < 8; v++)
                lw[(v + lh * 8) * 64 + n * 16 + lm] = (_Float16)p[n][v];
        __syncthreads();

        v16h ap[2];
#pragma unroll
        for (int ks = 0; ks < 2; ks++) {
            HFragU u;
            u.u4[0] = *reinterpret_cast<const uint4*>(lw + lm * 64 + ks * 32 + lh * 8);
            u.u4[1] = *reinterpret_cast<const uint4*>(lw + lm * 64 + ks * 32 + lh * 8 + 16);
            ap[ks] = u.h;
        }
        __syncthreads();

        // O += P @ V
#pragma unroll
        for (int ks = 0; ks < 2; ks++)
#pragma unroll
            for (int j = 0; j < 4; j++) {
                v16h bv = load_b_frag(Vp, 1024, j * 16 + lm, kc * 64 + ks * 32 + lh * 8);
                o[j] = __builtin_amdgcn_wmma_f32_16x16x32_f16(
                    false, ap[ks], false, bv, (short)0, o[j], false, false);
            }
    }

    float inv[8];
#pragma unroll
    for (int v = 0; v < 8; v++) inv[v] = 1.0f / l8[v];
#pragma unroll
    for (int j = 0; j < 4; j++)
#pragma unroll
        for (int v = 0; v < 8; v++) {
            int r  = qbase + v + lh * 8;
            int cc = h * 64 + j * 16 + lm;
            Out[((size_t)(b * 1024 + r)) * 512 + cc] = (_Float16)(o[j][v] * inv[v]);
        }
}

// ---------------------------------------------------------------------------
// Kernel 6: epilogue — +bo, +residual, layernorm over d=512, transpose out.
// One block per (b,s) row.
// ---------------------------------------------------------------------------
__global__ __launch_bounds__(256) void ln_kernel(const float* __restrict__ P,
                                                 const float* __restrict__ R,
                                                 const float* __restrict__ bo,
                                                 const float* __restrict__ gamma,
                                                 const float* __restrict__ beta,
                                                 float* __restrict__ out) {
    __shared__ float yrow[512];
    __shared__ float red[16];
    int row = blockIdx.x;
    int b = row >> 10, s = row & 1023;
    int t = threadIdx.x;

    float sum = 0.0f, sq = 0.0f;
    for (int c = t; c < 512; c += 256) {
        float y = P[(size_t)row * 512 + c] + bo[c] + R[(size_t)row * 512 + c];
        yrow[c] = y;
        sum += y;
        sq += y * y;
    }
#pragma unroll
    for (int msk = 1; msk < 32; msk <<= 1) {
        sum += __shfl_xor(sum, msk, 32);
        sq  += __shfl_xor(sq, msk, 32);
    }
    int wid = t >> 5, lane = t & 31;
    if (lane == 0) { red[wid] = sum; red[8 + wid] = sq; }
    __syncthreads();
    if (t == 0) {
        float S = 0.0f, Q2 = 0.0f;
        for (int i = 0; i < 8; i++) { S += red[i]; Q2 += red[8 + i]; }
        red[0] = S;
        red[8] = Q2;
    }
    __syncthreads();
    float mean = red[0] * (1.0f / 512.0f);
    float var  = red[8] * (1.0f / 512.0f) - mean * mean;
    float rstd = rsqrtf(var + 1e-5f);
    for (int c = t; c < 512; c += 256) {
        float yn = (yrow[c] - mean) * rstd * gamma[c] + beta[c];
        out[((size_t)(b * 512 + c)) * 1024 + s] = yn;
    }
}

// ---------------------------------------------------------------------------
extern "C" void kernel_launch(void* const* d_in, const int* in_sizes, int n_in,
                              void* d_out, int out_size, void* d_ws, size_t ws_size,
                              hipStream_t stream) {
    (void)in_sizes; (void)n_in; (void)out_size; (void)ws_size;

    const float* x     = (const float*)d_in[0];
    const float* Wq    = (const float*)d_in[1];
    const float* Wk    = (const float*)d_in[2];
    const float* Wv    = (const float*)d_in[3];
    const float* Wo    = (const float*)d_in[4];
    const float* bo    = (const float*)d_in[5];
    const float* gamma = (const float*)d_in[6];
    const float* beta  = (const float*)d_in[7];

    char* ws = (char*)d_ws;
    size_t off = 0;
    const size_t M = 8192, D = 512;

    float* Xf = (float*)(ws + off);         off += M * D * sizeof(float);     // residual
    _Float16* Xh = (_Float16*)(ws + off);   off += M * D * sizeof(_Float16);  // f16 input
    _Float16* attn = Xh;                    // reused after QKV GEMMs
    _Float16* Wqh = (_Float16*)(ws + off);  off += D * D * sizeof(_Float16);
    _Float16* Wkh = (_Float16*)(ws + off);  off += D * D * sizeof(_Float16);
    _Float16* Wvh = (_Float16*)(ws + off);  off += D * D * sizeof(_Float16);
    _Float16* Woh = (_Float16*)(ws + off);  off += D * D * sizeof(_Float16);
    _Float16* Qb  = (_Float16*)(ws + off);  off += M * D * sizeof(_Float16);
    _Float16* Kb  = (_Float16*)(ws + off);  off += M * D * sizeof(_Float16);
    float* proj = (float*)Qb;               // overlays dead Q+K (exactly M*D floats)
    _Float16* Vb  = (_Float16*)(ws + off);  off += M * D * sizeof(_Float16);
    _Float16* Vtb = (_Float16*)(ws + off);  off += M * D * sizeof(_Float16);

    prep_x_kernel<<<16384, 256, 0, stream>>>(x, Xf, Xh);
    conv_w_kernel<<<1024, 256, 0, stream>>>(Wq, Wk, Wv, Wo, Wqh, Wkh, Wvh, Woh);

    dim3 gg(64, 4);   // M/128 x N/128
    gemm_nt_kernel<true><<<gg, 256, 0, stream>>>(Xh, Wqh, Qb, 512, 512);
    gemm_nt_kernel<true><<<gg, 256, 0, stream>>>(Xh, Wkh, Kb, 512, 512);
    gemm_nt_kernel<true><<<gg, 256, 0, stream>>>(Xh, Wvh, Vb, 512, 512);

    transpose_v_kernel<<<16384, 256, 0, stream>>>(Vb, Vtb);

    attn_kernel<<<dim3(8, 64), 256, 0, stream>>>(Qb, Kb, Vtb, attn);

    gemm_nt_kernel<false><<<gg, 256, 0, stream>>>(attn, Woh, proj, 512, 512);

    ln_kernel<<<8192, 256, 0, stream>>>(proj, Xf, bo, gamma, beta, (float*)d_out);
}
